// MoNet_83906481094706
// MI455X (gfx1250) — matrix operand order repaired and checked by reading, not compiled, not run on hardware
//
#include <hip/hip_runtime.h>
#include <hip/hip_bf16.h>

// ---------------------------------------------------------------------------
// MoNet (GMM-conv x3 + pool + MLP head) for gfx1250 (MI455X), wave32 + WMMA.
// GEMM phase: f16 WMMA 16x16x32, fp32 accumulate. Edge phase: fp32, L2-bound.
// ---------------------------------------------------------------------------

#define NN  50000
#define EE  400000
#define KK  4
#define GG  64
#define EPSV 1e-15f
#define BN_EPS 1e-5f

typedef __attribute__((ext_vector_type(16))) _Float16 v16h;
typedef __attribute__((ext_vector_type(8)))  float    v8f;

// ---------------- utility ----------------
__global__ void kzero(float* __restrict__ p, int n) {
  int t = blockIdx.x * blockDim.x + threadIdx.x;
  if (t < n) p[t] = 0.0f;
}

// ---------------- pack A (fp32 [rows, ci] -> f16 WMMA A-tile layout) --------
// Tile = 16 rows x 32 K. Per-lane 16 halves contiguous (the exact operand
// vector for v_wmma_f32_16x16x32_f16). Lanes 0-15: M=0..15, K-halves 0-7 &
// 16-23; lanes 16-31: same rows, K 8-15 & 24-31 (ISA 7.12.2).
__global__ void pack_a_k(const float* __restrict__ src, _Float16* __restrict__ dst,
                         int rows, int ci) {
  int tid = blockIdx.x * blockDim.x + threadIdx.x;
  int total = rows * ci;
  if (tid >= total) return;
  int KT = ci >> 5;
  int tile   = tid >> 9;      // 512 halves per tile
  int within = tid & 511;
  int l  = within >> 4;       // lane 0..31
  int jj = within & 15;       // half index within lane vector
  int j = jj >> 1, r = jj & 1;
  int mt = tile / KT, kt = tile % KT;
  int row = mt * 16 + (l & 15);
  int kh  = l >> 4;
  int k = kt * 32 + ((j < 4) ? (kh * 8 + j * 2) : (16 + kh * 8 + (j - 4) * 2)) + r;
  dst[tid] = (_Float16)src[(size_t)row * ci + k];
}

// ---------------- pack B ([g | root] fp32 -> f16 WMMA B-tile layout) --------
// B tile = 32 K x 16 N. Lanes 0-15: N=0..15 with K 0-7 & 16-23 across the 8 VGPRs;
// lanes 16-31: K 8-15 & 24-31. Column n < 4*co comes from g, else from root.
__global__ void pack_b_k(const float* __restrict__ g, const float* __restrict__ root,
                         _Float16* __restrict__ dst, int ci, int co) {
  int ncols = 5 * co;
  int tid = blockIdx.x * blockDim.x + threadIdx.x;
  int total = ci * ncols;
  if (tid >= total) return;
  int KT = ci >> 5;
  int tile   = tid >> 9;
  int within = tid & 511;
  int l  = within >> 4;
  int jj = within & 15;
  int j = jj >> 1, r = jj & 1;
  int nt = tile / KT, kt = tile % KT;
  int col  = nt * 16 + (l & 15);
  int kg   = l >> 4;
  int krow = kt * 32 + ((j < 4) ? (kg * 8 + j * 2) : (16 + kg * 8 + (j - 4) * 2)) + r;
  float v = (col < 4 * co) ? g[(size_t)krow * (4 * co) + col]
                           : root[(size_t)krow * co + (col - 4 * co)];
  dst[tid] = (_Float16)v;
}

// ---------------- WMMA GEMM: one wave per 16x16 output tile -----------------
__global__ __launch_bounds__(32)
void wmma_gemm_k(const _Float16* __restrict__ A, const _Float16* __restrict__ B,
                 float* __restrict__ C, int KT, int ncols) {
  int mt = blockIdx.x, nt = blockIdx.y;
  int lane = threadIdx.x;                 // wave32, EXEC all ones
  v8f c = {};
  const _Float16* ap = A + ((size_t)mt * KT) * 512 + lane * 16;
  const _Float16* bp = B + ((size_t)nt * KT) * 512 + lane * 16;
  for (int kt = 0; kt < KT; ++kt) {
    v16h a = *(const v16h*)ap;  ap += 512;
    v16h b = *(const v16h*)bp;  bp += 512;
    c = __builtin_amdgcn_wmma_f32_16x16x32_f16(false, a, false, b,
                                               (short)0, c, false, false);
  }
  // C layout: VGPR r -> row (r + 8*(lane/16)), col = lane%16 (ISA 7.12.2)
  int col = nt * 16 + (lane & 15);
  int row = mt * 16 + ((lane >> 4) << 3);
#pragma unroll
  for (int r = 0; r < 8; ++r)
    C[(size_t)(row + r) * ncols + col] = c[r];
}

// ---------------- per-edge Gaussian mixture weights [E,4] -------------------
__global__ void edge_w_k(const float* __restrict__ ea, const float* __restrict__ mu,
                         const float* __restrict__ sigma, float* __restrict__ wbuf,
                         int e_) {
  int tid = blockIdx.x * blockDim.x + threadIdx.x;
  if (tid >= e_ * 4) return;
  int e = tid >> 2, k = tid & 3;
  float p0 = ea[e * 2], p1 = ea[e * 2 + 1];
  float d0 = p0 - mu[k * 2],     d1 = p1 - mu[k * 2 + 1];
  float s0 = sigma[k * 2],       s1 = sigma[k * 2 + 1];
  wbuf[tid] = __expf(-0.5f * (d0 * d0 / (EPSV + s0 * s0) +
                              d1 * d1 / (EPSV + s1 * s1)));
}

// ---------------- edge counts (mean aggregation denominator) ----------------
__global__ void count_k(const int* __restrict__ dst, float* __restrict__ cnt, int e_) {
  int e = blockIdx.x * blockDim.x + threadIdx.x;
  if (e < e_) atomicAdd(&cnt[dst[e]], 1.0f);
}

// ---------------- gather + weighted sum + scatter-add -----------------------
__global__ void edge_msg_k(const float* __restrict__ xt, const float* __restrict__ wbuf,
                           const int* __restrict__ src, const int* __restrict__ dst,
                           float* __restrict__ agg, int e_, int coShift) {
  int tid = blockIdx.x * blockDim.x + threadIdx.x;
  int total = e_ << coShift;
  if (tid >= total) return;
  int e  = tid >> coShift;
  int co_ = 1 << coShift;
  int co = tid & (co_ - 1);
  int ldn = 5 * co_;
  int s = src[e], d = dst[e];
  const float* xr = xt + (size_t)s * ldn + co;
  const float* w  = wbuf + (size_t)e * 4;
  float msg = xr[0]        * w[0] + xr[co_]     * w[1]
            + xr[2 * co_]  * w[2] + xr[3 * co_] * w[3];
  atomicAdd(&agg[(size_t)d * co_ + co], msg);
}

// ---------------- mean + root + bias + BN + ELU -----------------------------
__global__ void finalize_k(const float* __restrict__ agg, const float* __restrict__ cnt,
                           const float* __restrict__ xt, const float* __restrict__ bias,
                           const float* __restrict__ bng, const float* __restrict__ bnb,
                           const float* __restrict__ bnm, const float* __restrict__ bnv,
                           float* __restrict__ out, int n_, int coShift) {
  int tid = blockIdx.x * blockDim.x + threadIdx.x;
  int total = n_ << coShift;
  if (tid >= total) return;
  int n = tid >> coShift;
  int co_ = 1 << coShift;
  int co = tid & (co_ - 1);
  int ldn = 5 * co_;
  float c = cnt[n]; c = c < 1.0f ? 1.0f : c;
  float v = agg[tid] / c + xt[(size_t)n * ldn + 4 * co_ + co] + bias[co];
  float scale = bng[co] * rsqrtf(bnv[co] + BN_EPS);
  v = (v - bnm[co]) * scale + bnb[co];
  out[tid] = v > 0.0f ? v : (__expf(v) - 1.0f);
}

// ---------------- global mean pool (segment sum over batch) -----------------
__global__ void pool_k(const float* __restrict__ h, const int* __restrict__ batch,
                       float* __restrict__ pooled, float* __restrict__ gcnt, int n_) {
  int tid = blockIdx.x * blockDim.x + threadIdx.x;
  if (tid >= n_ * 64) return;
  int n = tid >> 6, co = tid & 63;
  int b = batch[n];
  atomicAdd(&pooled[b * 64 + co], h[tid]);
  if (co == 0) atomicAdd(&gcnt[b], 1.0f);
}

// ---------------- MLP head + log_softmax, one block per graph ---------------
__global__ __launch_bounds__(128)
void head_k(const float* __restrict__ pooled, const float* __restrict__ gcnt,
            const float* __restrict__ w1, const float* __restrict__ b1,
            const float* __restrict__ w2, const float* __restrict__ b2,
            float* __restrict__ out) {
  __shared__ float pm[64], z1[80], z2[16], mx, lse;
  int g = blockIdx.x, t = threadIdx.x;
  if (t < 64) {
    float c = gcnt[g]; c = c < 1.0f ? 1.0f : c;
    pm[t] = pooled[g * 64 + t] / c;
  }
  __syncthreads();
  if (t < 80) {
    float acc = b1[t];
    for (int i = 0; i < 64; ++i) acc += pm[i] * w1[i * 80 + t];
    z1[t] = acc > 0.0f ? acc : (__expf(acc) - 1.0f);
  }
  __syncthreads();
  if (t < 10) {
    float acc = b2[t];
    for (int i = 0; i < 80; ++i) acc += z1[i] * w2[i * 10 + t];
    z2[t] = acc > 0.0f ? acc : (__expf(acc) - 1.0f);
  }
  __syncthreads();
  if (t == 0) {
    float m = z2[0];
    for (int i = 1; i < 10; ++i) m = fmaxf(m, z2[i]);
    float s = 0.0f;
    for (int i = 0; i < 10; ++i) s += __expf(z2[i] - m);
    mx = m; lse = __logf(s);
  }
  __syncthreads();
  if (t < 10) out[g * 10 + t] = z2[t] - mx - lse;
}

// ===========================================================================
extern "C" void kernel_launch(void* const* d_in, const int* in_sizes, int n_in,
                              void* d_out, int out_size, void* d_ws, size_t ws_size,
                              hipStream_t stream) {
  (void)in_sizes; (void)n_in; (void)out_size; (void)ws_size;
  const float* x    = (const float*)d_in[0];
  const float* ea   = (const float*)d_in[1];
  const int*   ei   = (const int*)d_in[2];
  const int*   batch= (const int*)d_in[3];
  const int* srcI = ei;
  const int* dstI = ei + EE;

  const float *gW[3], *muW[3], *sgW[3], *rtW[3], *biW[3];
  const float *bnG[3], *bnB[3], *bnM[3], *bnV[3];
  for (int l = 0; l < 3; ++l) {
    int base = 4 + l * 9;
    gW[l]  = (const float*)d_in[base + 0];
    muW[l] = (const float*)d_in[base + 1];
    sgW[l] = (const float*)d_in[base + 2];
    rtW[l] = (const float*)d_in[base + 3];
    biW[l] = (const float*)d_in[base + 4];
    bnG[l] = (const float*)d_in[base + 5];
    bnB[l] = (const float*)d_in[base + 6];
    bnM[l] = (const float*)d_in[base + 7];
    bnV[l] = (const float*)d_in[base + 8];
  }
  const float* fc1w = (const float*)d_in[31];
  const float* fc1b = (const float*)d_in[32];
  const float* fc2w = (const float*)d_in[33];
  const float* fc2b = (const float*)d_in[34];
  float* outp = (float*)d_out;

  // ---- carve workspace ----
  unsigned char* base = (unsigned char*)d_ws;
  size_t off = 0;
  auto carve = [&](size_t bytes) -> void* {
    void* p = base + off;
    off += (bytes + 255) & ~(size_t)255;
    return p;
  };
  float*    hA     = (float*)   carve((size_t)NN * 64 * 4);
  float*    hB     = (float*)   carve((size_t)NN * 64 * 4);
  float*    xt     = (float*)   carve((size_t)NN * 320 * 4);
  float*    agg    = (float*)   carve((size_t)NN * 64 * 4);
  float*    cnt    = (float*)   carve((size_t)NN * 4);
  _Float16* apack  = (_Float16*)carve((size_t)NN * 64 * 2);
  _Float16* bpack  = (_Float16*)carve((size_t)64 * 320 * 2);
  float*    wbuf   = (float*)   carve((size_t)EE * 4 * 4);
  float*    pooled = (float*)   carve((size_t)GG * 64 * 4);
  float*    gcnt   = (float*)   carve((size_t)GG * 4);

  const int TB = 256;
  auto blocks = [](long long n, int tb) { return (int)((n + tb - 1) / tb); };

  // edge counts (same for all layers) + pooled accumulators
  kzero<<<blocks(NN, TB), TB, 0, stream>>>(cnt, NN);
  kzero<<<blocks(GG * 64, TB), TB, 0, stream>>>(pooled, GG * 64);
  kzero<<<blocks(GG, TB), TB, 0, stream>>>(gcnt, GG);
  count_k<<<blocks(EE, TB), TB, 0, stream>>>(dstI, cnt, EE);

  const int CiA[3] = {32, 32, 64};
  const int CoA[3] = {32, 64, 64};
  const int CoS[3] = {5, 6, 6};

  const float* layerIn = x;
  float* ping[2] = {hA, hB};

  for (int l = 0; l < 3; ++l) {
    int Ci = CiA[l], Co = CoA[l], cs = CoS[l];
    int KT = Ci >> 5;
    int ncols = 5 * Co;
    float* layerOut = ping[l & 1];

    // 1) pack A (node features) and B ([g | root]) into WMMA operand layout
    pack_a_k<<<blocks((long long)NN * Ci, TB), TB, 0, stream>>>(layerIn, apack, NN, Ci);
    pack_b_k<<<blocks((long long)Ci * ncols, TB), TB, 0, stream>>>(gW[l], rtW[l], bpack, Ci, Co);

    // 2) xt[N, 5*Co] = in @ [g | root]   (f16 WMMA, f32 accumulate)
    dim3 grid(NN / 16, ncols / 16);
    wmma_gemm_k<<<grid, 32, 0, stream>>>(apack, bpack, xt, KT, ncols);

    // 3) per-edge Gaussian weights, zero accumulator, gather+scatter
    edge_w_k<<<blocks((long long)EE * 4, TB), TB, 0, stream>>>(ea, muW[l], sgW[l], wbuf, EE);
    kzero<<<blocks((long long)NN * Co, TB), TB, 0, stream>>>(agg, NN * Co);
    edge_msg_k<<<blocks((long long)EE << cs, TB), TB, 0, stream>>>(xt, wbuf, srcI, dstI, agg, EE, cs);

    // 4) mean + root + bias + BN + ELU
    finalize_k<<<blocks((long long)NN << cs, TB), TB, 0, stream>>>(
        agg, cnt, xt, biW[l], bnG[l], bnB[l], bnM[l], bnV[l], layerOut, NN, cs);

    layerIn = layerOut;
  }

  // global mean pool + MLP head + log_softmax
  pool_k<<<blocks((long long)NN * 64, TB), TB, 0, stream>>>(layerIn, batch, pooled, gcnt, NN);
  head_k<<<GG, 128, 0, stream>>>(pooled, gcnt, fc1w, fc1b, fc2w, fc2b, outp);
}
